// CKConv_40475771797689
// MI455X (gfx1250) — matrix operand order repaired and checked by reading, not compiled, not run on hardware
//
#include <hip/hip_runtime.h>

typedef __attribute__((ext_vector_type(16))) _Float16 v16h;
typedef __attribute__((ext_vector_type(8)))  float    v8f;

#define L_LEN  8192
#define HID    128
#define MROWS  4096          // B*Cin
#define KSTEP  64            // K per pipeline stage
#define LDSW   72            // padded row stride (fp16): 144 B = 36 banks -> conflict-free b128 reads

// LDS offset of a generic pointer to __shared__ (low 32 bits of flat LDS address)
__device__ __forceinline__ unsigned lds_off(const void* p) {
    return (unsigned)(unsigned long long)p;
}

// ---------------------------------------------------------------------------
// Kernel 1: SIREN kernel-generating MLP.
// h2T[h*L + l] = sin(30*( sin(30*(rel*w1+b1)) . w2[h,:] + b2[h] ))   (fp16, N-major)
// ---------------------------------------------------------------------------
__global__ __launch_bounds__(HID) void siren_h2(
    const float* __restrict__ w1, const float* __restrict__ b1,
    const float* __restrict__ w2, const float* __restrict__ b2,
    _Float16* __restrict__ h2T)
{
    const int l = blockIdx.x;
    const int h = threadIdx.x;
    __shared__ float h1s[HID];

    const float rel = -1.0f + (2.0f * (float)l) / (float)(L_LEN - 1);
    h1s[h] = __sinf(30.0f * (rel * w1[h] + b1[h]));
    __syncthreads();

    float acc = b2[h];
    const float* wr = w2 + (size_t)h * HID;
#pragma unroll 8
    for (int j = 0; j < HID; ++j) acc += h1s[j] * wr[j];

    h2T[(size_t)h * L_LEN + l] = (_Float16)__sinf(30.0f * acc);
}

// ---------------------------------------------------------------------------
// Kernel 2: G[4096,128] = X[4096,8192](fp32 -> fp16 in-register) @ h2[8192,128]
// + s[m] = row sums of X.
// 256 thr = 8 waves; tile 128(M) x 128(N); K-step 64, double-buffered LDS.
// A: global_load_b128 -> regs -> cvt_pk_f16 -> LDS (held across WMMAs).
// B: global_load_async_to_lds_b128 (ASYNCcnt) -- zero VGPR staging.
// Wave: 2 M-tiles x 4 N-tiles -> 16 v_wmma_f32_16x16x32_f16 per K-step.
// ---------------------------------------------------------------------------
__global__ __launch_bounds__(256)
__attribute__((amdgpu_waves_per_eu(2, 2)))
void ckconv_gemm(
    const float* __restrict__ x,      // [4096, 8192]
    const _Float16* __restrict__ h2T, // [128, 8192]
    float* __restrict__ G,            // [4096, 128]
    float* __restrict__ s)            // [4096]
{
    const int tid  = threadIdx.x;
    const int lane = tid & 31;
    const int wave = tid >> 5;           // 0..7
    const int m0   = blockIdx.x * 128;   // 32 blocks

    __shared__ _Float16 As[2][128 * LDSW];   // 2 x 18 KB
    __shared__ _Float16 Bs[2][128 * LDSW];   // 2 x 18 KB  (72 KB total)

    // wave -> 2 M-tiles x 4 N-tiles
    const int mbase = (wave & 3) * 32;
    const int nbase = (wave >> 2) * 64;

    v8f acc[2][4];
#pragma unroll
    for (int mi = 0; mi < 2; ++mi)
#pragma unroll
        for (int nt = 0; nt < 4; ++nt)
#pragma unroll
            for (int r = 0; r < 8; ++r) acc[mi][nt][r] = 0.f;

    float sumA[8];
#pragma unroll
    for (int rep = 0; rep < 8; ++rep) sumA[rep] = 0.f;

    // per-thread load coordinates (fixed across K-steps)
    const int arow = tid >> 4;           // 0..15  (+ rep*16 -> rows 0..127)
    const int ac4  = (tid & 15) * 4;     // 0..60  (float4 column)
    const int bn   = tid >> 1;           // 0..127 (h2 row = N index)
    const int bkh  = (tid & 1) * 32;     // 0 / 32 (k offset)

    float4 ra[8];

    auto loadA = [&](int k0) {
#pragma unroll
        for (int rep = 0; rep < 8; ++rep)
            ra[rep] = *(const float4*)(x + (size_t)(m0 + rep * 16 + arow) * L_LEN + k0 + ac4);
    };

    // B tile: 64 B per thread straight into LDS, no VGPR staging (ASYNCcnt).
    // INST_OFFSET is added to BOTH the LDS and the global address (ISA 08 §4.4).
    auto asyncB = [&](int k0, int buf) {
        const unsigned  dst = lds_off(&Bs[buf][bn * LDSW + bkh]);
        const _Float16* src = h2T + (size_t)bn * L_LEN + k0 + bkh;
        asm volatile(
            "global_load_async_to_lds_b128 %0, %1, off\n\t"
            "global_load_async_to_lds_b128 %0, %1, off offset:16\n\t"
            "global_load_async_to_lds_b128 %0, %1, off offset:32\n\t"
            "global_load_async_to_lds_b128 %0, %1, off offset:48"
            :: "v"(dst), "v"(src) : "memory");
    };

    auto storeA = [&](int buf) {
#pragma unroll
        for (int rep = 0; rep < 8; ++rep) {
            const float4 f = ra[rep];
            sumA[rep] += f.x + f.y + f.z + f.w;               // free row-sum for s[]
            _Float16* dst = &As[buf][(rep * 16 + arow) * LDSW + ac4];
            dst[0] = (_Float16)f.x; dst[1] = (_Float16)f.y;
            dst[2] = (_Float16)f.z; dst[3] = (_Float16)f.w;
        }
    };

    auto compute = [&](int buf) {
#pragma unroll
        for (int ks = 0; ks < 2; ++ks) {
            // A frag (16-bit A 16x32): lanes 0-15 K=0-7/16-23, lanes 16-31 K=8-15/24-31
            const int kfA = ks * 32 + (lane >> 4) * 8;
            v16h af[2];
#pragma unroll
            for (int mi = 0; mi < 2; ++mi) {
                const _Float16* ap = &As[buf][(mbase + mi * 16 + (lane & 15)) * LDSW + kfA];
#pragma unroll
                for (int t = 0; t < 8; ++t) { af[mi][t] = ap[t]; af[mi][8 + t] = ap[16 + t]; }
            }
            // B frag: lanes 0-15 K=0-7/16-23, lanes 16-31 K=8-15/24-31 (matches A)
            const int kfB = ks * 32 + (lane >> 4) * 8;
#pragma unroll
            for (int nt = 0; nt < 4; ++nt) {
                v16h bf;
                const _Float16* bp = &Bs[buf][(nbase + nt * 16 + (lane & 15)) * LDSW + kfB];
#pragma unroll
                for (int t = 0; t < 8; ++t) { bf[t] = bp[t]; bf[8 + t] = bp[16 + t]; }
                acc[0][nt] = __builtin_amdgcn_wmma_f32_16x16x32_f16(
                    false, af[0], false, bf, (short)0, acc[0][nt], false, false);
                acc[1][nt] = __builtin_amdgcn_wmma_f32_16x16x32_f16(
                    false, af[1], false, bf, (short)0, acc[1][nt], false, false);
            }
        }
    };

    // ---- prologue: stage tile 0 ----
    asyncB(0, 0);
    loadA(0);
    storeA(0);
    asm volatile("s_wait_asynccnt 0x0" ::: "memory");
    __syncthreads();

    // ---- pipeline: tile t+1 loads (global + async DMA) overlap WMMAs on tile t ----
    const int T = L_LEN / KSTEP;   // 128
    for (int t = 0; t < T; ++t) {
        const int cur = t & 1;
        if (t + 1 < T) {
            asyncB((t + 1) * KSTEP, 1 - cur);   // DMA straight into other buffer
            loadA((t + 1) * KSTEP);             // A loads in flight...
        }
        compute(cur);                           // ...hidden behind 16 WMMAs
        if (t + 1 < T) storeA(1 - cur);         // convert + stage other buffer
        asm volatile("s_wait_asynccnt 0x0" ::: "memory");
        __syncthreads();
    }

    // ---- write G (C/D layout: VGPR r -> M=r lanes<16, M=r+8 lanes>=16) ----
    const int rowb = m0 + mbase + (lane >> 4) * 8;
    const int nl   = lane & 15;
#pragma unroll
    for (int mi = 0; mi < 2; ++mi)
#pragma unroll
        for (int nt = 0; nt < 4; ++nt)
#pragma unroll
            for (int r = 0; r < 8; ++r)
                G[(size_t)(rowb + mi * 16 + r) * HID + nbase + nt * 16 + nl] = acc[mi][nt][r];

    // ---- reduce row sums -> s[] (rows unique to this block; no atomics) ----
    __syncthreads();
    float* red = (float*)As;
#pragma unroll
    for (int rep = 0; rep < 8; ++rep) {
        red[tid] = sumA[rep];
        __syncthreads();
        if ((tid & 15) == 0) {
            float v = 0.f;
#pragma unroll
            for (int j = 0; j < 16; ++j) v += red[tid + j];
            s[m0 + rep * 16 + (tid >> 4)] = v;
        }
        __syncthreads();
    }
}

// ---------------------------------------------------------------------------
// Kernel 3: out[b,o] = bias[o] + sum_i ( G[b,i,:].w3[o*64+i,:] + s[b,i]*b3[o*64+i] )
// ---------------------------------------------------------------------------
__global__ __launch_bounds__(64) void ckconv_final(
    const float* __restrict__ G, const float* __restrict__ s,
    const float* __restrict__ w3, const float* __restrict__ b3,
    const float* __restrict__ bias, float* __restrict__ out)
{
    const int b = blockIdx.x;
    const int o = threadIdx.x;

    __shared__ float Gs[64 * 128];   // 32 KB
    __shared__ float Ss[64];

    const float* gsrc = G + (size_t)b * 64 * 128;
    for (int idx = o; idx < 64 * 128 / 4; idx += 64)
        ((float4*)Gs)[idx] = ((const float4*)gsrc)[idx];
    Ss[o] = s[b * 64 + o];
    __syncthreads();

    float acc = bias[o];
    for (int i = 0; i < 64; ++i) {
        const float* wrow = w3 + (size_t)(o * 64 + i) * HID;
        const float* grow = &Gs[i * HID];
        float d = 0.f;
#pragma unroll 8
        for (int h = 0; h < HID; ++h) d += grow[h] * wrow[h];
        acc += d + Ss[i] * b3[o * 64 + i];
    }
    out[b * 64 + o] = acc;
}

// ---------------------------------------------------------------------------
extern "C" void kernel_launch(void* const* d_in, const int* in_sizes, int n_in,
                              void* d_out, int out_size, void* d_ws, size_t ws_size,
                              hipStream_t stream)
{
    (void)in_sizes; (void)n_in; (void)out_size; (void)ws_size;

    const float* x    = (const float*)d_in[0];   // [64,64,8192]
    const float* w1   = (const float*)d_in[1];   // [128,1]
    const float* b1   = (const float*)d_in[2];   // [128]
    const float* w2   = (const float*)d_in[3];   // [128,128]
    const float* b2   = (const float*)d_in[4];   // [128]
    const float* w3   = (const float*)d_in[5];   // [4096,128]
    const float* b3   = (const float*)d_in[6];   // [4096]
    const float* bias = (const float*)d_in[7];   // [64]
    float* out = (float*)d_out;                  // [64,64,1]

    char* ws = (char*)d_ws;
    _Float16* h2T = (_Float16*)(ws);                        // 2 MB
    float*    G   = (float*)(ws + (size_t)2 * 1024 * 1024); // 2 MB
    float*    s   = (float*)(ws + (size_t)4 * 1024 * 1024); // 16 KB

    siren_h2    <<<L_LEN,       HID, 0, stream>>>(w1, b1, w2, b2, h2T);
    ckconv_gemm <<<MROWS / 128, 256, 0, stream>>>(x, h2T, G, s);
    ckconv_final<<<64,           64, 0, stream>>>(G, s, w3, b3, bias, out);
}
